// RouterKT_19542101196821
// MI455X (gfx1250) — matrix operand order repaired and checked by reading, not compiled, not run on hardware
//
#include <hip/hip_runtime.h>
#include <hip/hip_bf16.h>

#define BS 32
#define SEQ 512
#define DMODEL 512
#define NH 8
#define HS 2
#define HSEL 4
#define NL 6
#define DK 64
#define HD (NH - HS) /* 6 dynamic heads */

typedef __attribute__((ext_vector_type(16))) _Float16 v16h;
typedef __attribute__((ext_vector_type(8)))  _Float16 v8h;
typedef __attribute__((ext_vector_type(8)))  float    v8f;

// ---------------------------------------------------------------------------
// WMMA fragment loaders (f16, 16x16x32, wave32; this code's contiguous k order, same for A and B)
// A (16x32 MxK): lane<16 holds M=lane, K=k..k+15 contiguous; lane>=16 holds
//                M=lane-16, K=k+16..k+31.
// B (32x16 KxN): lane<16 holds N=lane, K=k..k+15 contiguous; lane>=16 holds
//                N=lane-16, K=k+16..k+31. With W stored [N,K] row-major the
//                B-frag is a single contiguous 32B load.
// ---------------------------------------------------------------------------
__device__ __forceinline__ v16h load_a_frag(const _Float16* A, int lda, int row,
                                            int k, int lane) {
  const int hh = (lane >> 4) & 1;
  const _Float16* p = A + (size_t)row * lda + k + hh * 16;
  v8h lo = *(const v8h*)p;
  v8h hi = *(const v8h*)(p + 8);
  v16h r;
#pragma unroll
  for (int i = 0; i < 8; ++i) { r[i] = lo[i]; r[i + 8] = hi[i]; }
  return r;
}

__device__ __forceinline__ v16h load_b_frag(const _Float16* W, int ldk, int ncol,
                                            int k, int lane) {
  const int hh = (lane >> 4) & 1;
  return *(const v16h*)(W + (size_t)ncol * ldk + k + hh * 16);
}

// ---------------------------------------------------------------------------
// fp32 -> (fp32 copy?, f16) conversion
// ---------------------------------------------------------------------------
__global__ void cvt_kernel(const float* __restrict__ src, float* __restrict__ d32,
                           _Float16* __restrict__ d16, size_t n) {
  size_t i = (size_t)blockIdx.x * blockDim.x + threadIdx.x;
  if (i < n) {
    float v = src[i];
    if (d32) d32[i] = v;
    d16[i] = (_Float16)v;
  }
}

__global__ void zero_kernel(float* __restrict__ p, int n) {
  int i = blockIdx.x * blockDim.x + threadIdx.x;
  if (i < n) p[i] = 0.f;
}

// ---------------------------------------------------------------------------
// C[M,N] = A[M,K] @ W[N,K]^T + bias   (f16 in, f32 accumulate via WMMA)
// block = 256 threads = 8 waves; wave tile 32x64 (2 A-frags x 4 B-frags =
// 8 WMMAs per K step, ~21 f16-FLOP/byte); block tile 64x256.
// ---------------------------------------------------------------------------
template <bool RELU>
__global__ void __launch_bounds__(256)
gemm_bias_kernel(const _Float16* __restrict__ A, const _Float16* __restrict__ W,
                 const float* __restrict__ bias, float* __restrict__ Cf,
                 _Float16* __restrict__ Ch, int M, int N, int K) {
  const int lane = threadIdx.x & 31;
  const int wave = threadIdx.x >> 5;
  const int ln15 = lane & 15;
  const int hh   = (lane >> 4) & 1;
  const int bm = blockIdx.y * 64 + (wave >> 2) * 32;
  const int bn = blockIdx.x * 256 + (wave & 3) * 64;

  v8f acc[2][4] = {};
  for (int k = 0; k < K; k += 32) {
    __builtin_prefetch(A + (size_t)(bm + ln15) * K + k + 128, 0, 1);  // global_prefetch_b8
    v16h a0 = load_a_frag(A, K, bm + ln15, k, lane);
    v16h a1 = load_a_frag(A, K, bm + 16 + ln15, k, lane);
#pragma unroll
    for (int t = 0; t < 4; ++t) {
      v16h b = load_b_frag(W, K, bn + t * 16 + ln15, k, lane);
      acc[0][t] = __builtin_amdgcn_wmma_f32_16x16x32_f16(false, a0, false, b,
                                                         (short)0, acc[0][t], false, false);
      acc[1][t] = __builtin_amdgcn_wmma_f32_16x16x32_f16(false, a1, false, b,
                                                         (short)0, acc[1][t], false, false);
    }
  }

#pragma unroll
  for (int mh = 0; mh < 2; ++mh) {
    const int rbase = bm + mh * 16 + hh * 8;
#pragma unroll
    for (int t = 0; t < 4; ++t) {
      const int col = bn + t * 16 + ln15;
      const float bv = bias ? bias[col] : 0.f;
#pragma unroll
      for (int r = 0; r < 8; ++r) {
        float o = acc[mh][t][r] + bv;
        if (RELU) o = fmaxf(o, 0.f);
        if (Cf) Cf[(size_t)(rbase + r) * N + col] = o;
        if (Ch) Ch[(size_t)(rbase + r) * N + col] = (_Float16)o;
      }
    }
  }
}

// ---------------------------------------------------------------------------
// MoH attention (flash-style online softmax), q==k (kq_same), per (b,h).
// grid = (SEQ/64, BS*NH), block = 128 (4 waves, one 16-row query tile each).
// Key tile (32x64 f16) is staged once per block into LDS with CDNA5
// global_load_async_to_lds_b128 (ASYNCcnt), so all 4 waves share it; V tile
// is staged transposed for the ctx-GEMM B operand. strict -> tril(k=-1);
// global row 0 forced to 0 (uniform attention); ctx scaled by routing[b,h].
// ---------------------------------------------------------------------------
__global__ void __launch_bounds__(128)
moh_attention_kernel(const _Float16* __restrict__ q, const _Float16* __restrict__ v,
                     const float* __restrict__ routing, _Float16* __restrict__ ctx,
                     int strict) {
  __shared__ _Float16 Ks[32][64];     // key tile, row-major [key][dk]   (4 KB)
  __shared__ _Float16 Vt[64][32];     // transposed V tile [dk][key]     (4 KB)
  __shared__ _Float16 Ps[4][16][32];  // per-wave P staging (A-frag src) (4 KB)

  const int lane = threadIdx.x & 31;
  const int wave = threadIdx.x >> 5;
  const int ln15 = lane & 15;
  const int hh   = (lane >> 4) & 1;
  const int bh = blockIdx.y;
  const int b = bh / NH, h = bh % NH;
  const int qb = blockIdx.x * 64 + wave * 16;

  const _Float16* qbase = q + (size_t)b * SEQ * DMODEL + h * DK;
  const _Float16* vbase = v + (size_t)b * SEQ * DMODEL + h * DK;
  const unsigned ks_base = (unsigned)(size_t)&Ks[0][0];

  // async-staging geometry: 128 threads x 2 x 16B = 4KB key tile
  const int srow = threadIdx.x >> 3;         // 0..15 (key within tile)
  const int scol = (threadIdx.x & 7) * 8;    // 0..56 (dk, halves)

  const v16h qa0 = load_a_frag(qbase, DMODEL, qb + ln15, 0, lane);
  const v16h qa1 = load_a_frag(qbase, DMODEL, qb + ln15, 32, lane);

  v8f acc[4] = {};
  float m_run[8], l_run[8];
#pragma unroll
  for (int r = 0; r < 8; ++r) { m_run[r] = -3.0e38f; l_run[r] = 0.f; }

  for (int kb = 0; kb < SEQ; kb += 32) {
    __syncthreads();
    {  // CDNA5 async copy: global -> LDS, no VGPR staging, ASYNCcnt tracked
      unsigned voff0 = (unsigned)(((kb + srow) * DMODEL + scol) * 2);
      unsigned lds0 = ks_base + (unsigned)((srow * 64 + scol) * 2);
      asm volatile("global_load_async_to_lds_b128 %0, %1, %2"
                   :: "v"(lds0), "v"(voff0), "s"(qbase) : "memory");
      unsigned voff1 = (unsigned)(((kb + srow + 16) * DMODEL + scol) * 2);
      unsigned lds1 = ks_base + (unsigned)(((srow + 16) * 64 + scol) * 2);
      asm volatile("global_load_async_to_lds_b128 %0, %1, %2"
                   :: "v"(lds1), "v"(voff1), "s"(qbase) : "memory");
    }
    {  // cooperative transposed stage of V tile (32 keys x 64 dims)
      const int row = threadIdx.x >> 2;        // key within tile
      const int col = (threadIdx.x & 3) * 16;  // dim base
      v16h vv = *(const v16h*)(vbase + (size_t)(kb + row) * DMODEL + col);
#pragma unroll
      for (int j = 0; j < 16; ++j) Vt[col + j][row] = vv[j];
    }
    asm volatile("s_wait_asynccnt 0x0" ::: "memory");
    __syncthreads();

    // scores from LDS key tile: two 16x16 tiles, K=64 in two WMMA steps
    v8f sc[2] = {};
#pragma unroll
    for (int t = 0; t < 2; ++t) {
      v16h b0 = *(const v16h*)&Ks[t * 16 + ln15][hh * 16];
      sc[t] = __builtin_amdgcn_wmma_f32_16x16x32_f16(false, qa0, false, b0,
                                                     (short)0, sc[t], false, false);
      v16h b1 = *(const v16h*)&Ks[t * 16 + ln15][32 + hh * 16];
      sc[t] = __builtin_amdgcn_wmma_f32_16x16x32_f16(false, qa1, false, b1,
                                                     (short)0, sc[t], false, false);
    }

    // mask + online softmax per row (row = qb + hh*8 + r)
    float pr[2][8];
#pragma unroll
    for (int r = 0; r < 8; ++r) {
      const int grow = qb + hh * 8 + r;
      float s0 = sc[0][r] * 0.125f;
      float s1 = sc[1][r] * 0.125f;
      const int c0 = kb + ln15, c1 = kb + 16 + ln15;
      const int lim = grow - strict;
      if (grow == 0) { s0 = 0.f; s1 = 0.f; }  // pad-zero row: uniform attention
      else {
        if (c0 > lim) s0 = -1.0e9f;
        if (c1 > lim) s1 = -1.0e9f;
      }
      float mx = fmaxf(s0, s1);
#pragma unroll
      for (int off = 8; off >= 1; off >>= 1) mx = fmaxf(mx, __shfl_xor(mx, off, 32));
      const float mnew = fmaxf(m_run[r], mx);
      const float corr = __expf(m_run[r] - mnew);
      const float p0 = __expf(s0 - mnew);
      const float p1 = __expf(s1 - mnew);
      float ss = p0 + p1;
#pragma unroll
      for (int off = 8; off >= 1; off >>= 1) ss += __shfl_xor(ss, off, 32);
      l_run[r] = l_run[r] * corr + ss;
      m_run[r] = mnew;
#pragma unroll
      for (int t = 0; t < 4; ++t) acc[t][r] *= corr;
      pr[0][r] = p0; pr[1][r] = p1;
    }

    // restage P (16x32 f16) from C-layout into A-frag layout via LDS
#pragma unroll
    for (int t = 0; t < 2; ++t)
#pragma unroll
      for (int r = 0; r < 8; ++r)
        Ps[wave][hh * 8 + r][t * 16 + ln15] = (_Float16)pr[t][r];

    v16h pa;
    {
      const _Float16* pp = &Ps[wave][ln15][hh * 16];
      v8h lo = *(const v8h*)pp;
      v8h hi = *(const v8h*)(pp + 8);
#pragma unroll
      for (int i = 0; i < 8; ++i) { pa[i] = lo[i]; pa[i + 8] = hi[i]; }
    }
#pragma unroll
    for (int t = 0; t < 4; ++t) {  // ctx(16x64) += P(16x32) @ V(32x64)
      v16h bv = *(const v16h*)&Vt[t * 16 + ln15][hh * 16];
      acc[t] = __builtin_amdgcn_wmma_f32_16x16x32_f16(false, pa, false, bv,
                                                      (short)0, acc[t], false, false);
    }
  }

  const float rscale = routing[b * NH + h];
#pragma unroll
  for (int t = 0; t < 4; ++t)
#pragma unroll
    for (int r = 0; r < 8; ++r) {
      const int row = qb + hh * 8 + r;
      const float o = acc[t][r] / l_run[r] * rscale;
      ctx[((size_t)b * SEQ + row) * DMODEL + h * DK + t * 16 + ln15] = (_Float16)o;
    }
}

// ---------------------------------------------------------------------------
// routing: gates = softmax(q @ Wg^T) over 6 dyn heads; top-4 (drop 2 smallest);
// accumulate per-(b,head) routing sums + f/P balance statistics.
// ---------------------------------------------------------------------------
__global__ void routing_kernel(const float* __restrict__ q, const float* __restrict__ Wg,
                               float* __restrict__ routing, float* __restrict__ fP) {
  const int tok = blockIdx.x * blockDim.x + threadIdx.x;
  if (tok >= BS * SEQ) return;
  const float* qr = q + (size_t)tok * DMODEL;
  float g[HD];
#pragma unroll
  for (int j = 0; j < HD; ++j) g[j] = 0.f;
  for (int k = 0; k < DMODEL; ++k) {
    const float qv = qr[k];
#pragma unroll
    for (int j = 0; j < HD; ++j) g[j] += qv * Wg[j * DMODEL + k];
  }
  float mx = g[0];
#pragma unroll
  for (int j = 1; j < HD; ++j) mx = fmaxf(mx, g[j]);
  float se = 0.f;
#pragma unroll
  for (int j = 0; j < HD; ++j) { g[j] = __expf(g[j] - mx); se += g[j]; }
  const float inv = 1.f / se;
#pragma unroll
  for (int j = 0; j < HD; ++j) g[j] *= inv;

  int d0 = 0; float dmin = g[0];
#pragma unroll
  for (int j = 1; j < HD; ++j) if (g[j] < dmin) { dmin = g[j]; d0 = j; }
  int d1 = -1; dmin = 3.0e38f;
#pragma unroll
  for (int j = 0; j < HD; ++j) if (j != d0 && g[j] < dmin) { dmin = g[j]; d1 = j; }

  const int bidx = tok / SEQ;
#pragma unroll
  for (int j = 0; j < HD; ++j) {
    atomicAdd(&fP[HD + j], g[j]);  // P: sum of gates
    if (j != d0 && j != d1) {
      atomicAdd(&routing[bidx * NH + HS + j], g[j]);
      atomicAdd(&fP[j], 1.f);      // f: selection counts
    }
  }
}

__global__ void routing_finalize_kernel(float* __restrict__ routing,
                                        const float* __restrict__ fP,
                                        float* __restrict__ bal) {
  const int i = blockIdx.x * blockDim.x + threadIdx.x;
  if (i < BS * NH) {
    const int h = i % NH;
    routing[i] = (h < HS) ? 1.f : routing[i] * (1.f / SEQ);
  }
  if (i == 0) {
    const float ntok = (float)(BS * SEQ);
    float fsum = 0.f, psum = 0.f;
    for (int j = 0; j < HD; ++j) { fsum += fP[j]; psum += fP[HD + j] / ntok; }
    float bb = 0.f;
    for (int j = 0; j < HD; ++j)
      bb += (fP[j] / (fsum + 1e-5f)) * ((fP[HD + j] / ntok) / (psum + 1e-5f));
    *bal = bb;
  }
}

// ---------------------------------------------------------------------------
// out = LayerNorm(x + res) * g + b ; writes fp32 + f16. One wave per row.
// ---------------------------------------------------------------------------
__global__ void __launch_bounds__(256)
add_ln_kernel(const float* __restrict__ x, const float* __restrict__ res,
              const float* __restrict__ g, const float* __restrict__ bta,
              float* __restrict__ o32, _Float16* __restrict__ o16) {
  const int row = blockIdx.x * 8 + (threadIdx.x >> 5);
  const int lane = threadIdx.x & 31;
  const float* xr = x + (size_t)row * DMODEL;
  const float* rr = res + (size_t)row * DMODEL;
  float vbuf[16];
  float s = 0.f;
#pragma unroll
  for (int i = 0; i < 16; ++i) {
    const float t = xr[lane + i * 32] + rr[lane + i * 32];
    vbuf[i] = t; s += t;
  }
#pragma unroll
  for (int off = 16; off >= 1; off >>= 1) s += __shfl_xor(s, off, 32);
  const float mean = s * (1.f / DMODEL);
  float vs = 0.f;
#pragma unroll
  for (int i = 0; i < 16; ++i) { const float d = vbuf[i] - mean; vs += d * d; }
#pragma unroll
  for (int off = 16; off >= 1; off >>= 1) vs += __shfl_xor(vs, off, 32);
  const float rstd = rsqrtf(vs * (1.f / DMODEL) + 1e-5f);
#pragma unroll
  for (int i = 0; i < 16; ++i) {
    const int c = lane + i * 32;
    const float o = (vbuf[i] - mean) * rstd * g[c] + bta[c];
    o32[(size_t)row * DMODEL + c] = o;
    o16[(size_t)row * DMODEL + c] = (_Float16)o;
  }
}

// ---------------------------------------------------------------------------
extern "C" void kernel_launch(void* const* d_in, const int* in_sizes, int n_in,
                              void* d_out, int out_size, void* d_ws, size_t ws_size,
                              hipStream_t stream) {
  (void)in_sizes; (void)n_in; (void)out_size; (void)ws_size;
  const float* q_embed = (const float*)d_in[0];
  const float* qa_embed = (const float*)d_in[1];
  const float* Wq = (const float*)d_in[2];
  const float* bq = (const float*)d_in[3];
  const float* Wv = (const float*)d_in[4];
  const float* bv = (const float*)d_in[5];
  const float* Wg = (const float*)d_in[6];
  const float* Wo = (const float*)d_in[7];
  const float* bo = (const float*)d_in[8];
  const float* ln1_g = (const float*)d_in[9];
  const float* ln1_b = (const float*)d_in[10];
  const float* W1 = (const float*)d_in[11];
  const float* b1 = (const float*)d_in[12];
  const float* W2 = (const float*)d_in[13];
  const float* b2 = (const float*)d_in[14];
  const float* ln2_g = (const float*)d_in[15];
  const float* ln2_b = (const float*)d_in[16];

  const size_t NE = (size_t)BS * SEQ * DMODEL;        // 8,388,608
  const size_t WE = (size_t)NL * DMODEL * DMODEL;     // 1,572,864 per weight

  char* ws = (char*)d_ws;
  size_t off = 0;
  auto alloc = [&](size_t bytes) -> void* {
    void* p = (void*)(ws + off);
    off = (off + bytes + 255) & ~(size_t)255;
    return p;
  };

  _Float16* hWq = (_Float16*)alloc(WE * 2);
  _Float16* hWv = (_Float16*)alloc(WE * 2);
  _Float16* hWo = (_Float16*)alloc(WE * 2);
  _Float16* hW1 = (_Float16*)alloc(WE * 2);
  _Float16* hW2 = (_Float16*)alloc(WE * 2);
  float* x32 = (float*)alloc(NE * 4);
  float* y32 = (float*)alloc(NE * 4);
  float* q32 = (float*)alloc(NE * 4);
  float* t32 = (float*)alloc(NE * 4);
  float* h32 = (float*)alloc(NE * 4);
  _Float16* x16  = (_Float16*)alloc(NE * 2);
  _Float16* y16  = (_Float16*)alloc(NE * 2);
  _Float16* q16  = (_Float16*)alloc(NE * 2);
  _Float16* v16_ = (_Float16*)alloc(NE * 2);
  _Float16* c16  = (_Float16*)alloc(NE * 2);
  _Float16* h16  = (_Float16*)alloc(NE * 2);
  _Float16* f16b = (_Float16*)alloc(NE * 2);
  float* routing = (float*)alloc((BS * NH + 2 * HD) * 4);
  float* fP = routing + BS * NH;

  float* out32 = (float*)d_out;
  float* bal = out32 + NE;

  const int M = BS * SEQ;
  const dim3 cvw((unsigned)((WE + 255) / 256)), cvn((unsigned)((NE + 255) / 256));
  cvt_kernel<<<cvw, 256, 0, stream>>>(Wq, nullptr, hWq, WE);
  cvt_kernel<<<cvw, 256, 0, stream>>>(Wv, nullptr, hWv, WE);
  cvt_kernel<<<cvw, 256, 0, stream>>>(Wo, nullptr, hWo, WE);
  cvt_kernel<<<cvw, 256, 0, stream>>>(W1, nullptr, hW1, WE);
  cvt_kernel<<<cvw, 256, 0, stream>>>(W2, nullptr, hW2, WE);
  cvt_kernel<<<cvn, 256, 0, stream>>>(q_embed, x32, x16, NE);
  cvt_kernel<<<cvn, 256, 0, stream>>>(qa_embed, y32, y16, NE);

  auto run_layer = [&](int i, float* xq32, const _Float16* xq16, const _Float16* xv16,
                       int strict, float* o32, _Float16* o16) {
    const size_t wo = (size_t)i * DMODEL * DMODEL;
    const dim3 gg(DMODEL / 256, M / 64);  // (2, 256)
    gemm_bias_kernel<false><<<gg, 256, 0, stream>>>(xq16, hWq + wo, bq + i * DMODEL,
                                                    q32, q16, M, DMODEL, DMODEL);
    gemm_bias_kernel<false><<<gg, 256, 0, stream>>>(xv16, hWv + wo, bv + i * DMODEL,
                                                    nullptr, v16_, M, DMODEL, DMODEL);
    zero_kernel<<<dim3(2), 256, 0, stream>>>(routing, BS * NH + 2 * HD);
    routing_kernel<<<dim3(M / 256), 256, 0, stream>>>(q32, Wg + (size_t)i * HD * DMODEL,
                                                      routing, fP);
    routing_finalize_kernel<<<dim3(1), 256, 0, stream>>>(routing, fP, bal);
    moh_attention_kernel<<<dim3(SEQ / 64, BS * NH), 128, 0, stream>>>(q16, v16_, routing,
                                                                      c16, strict);
    gemm_bias_kernel<false><<<gg, 256, 0, stream>>>(c16, hWo + wo, bo + i * DMODEL,
                                                    t32, nullptr, M, DMODEL, DMODEL);
    add_ln_kernel<<<dim3(M / 8), 256, 0, stream>>>(xq32, t32, ln1_g + i * DMODEL,
                                                   ln1_b + i * DMODEL, h32, h16);
    gemm_bias_kernel<true><<<gg, 256, 0, stream>>>(h16, hW1 + wo, b1 + i * DMODEL,
                                                   nullptr, f16b, M, DMODEL, DMODEL);
    gemm_bias_kernel<false><<<gg, 256, 0, stream>>>(f16b, hW2 + wo, b2 + i * DMODEL,
                                                    t32, nullptr, M, DMODEL, DMODEL);
    add_ln_kernel<<<dim3(M / 8), 256, 0, stream>>>(h32, t32, ln2_g + i * DMODEL,
                                                   ln2_b + i * DMODEL, o32, o16);
  };

  // blocks_1: self-attention on y (mask=1)
  run_layer(0, y32, y16, y16, 0, y32, y16);
  run_layer(1, y32, y16, y16, 0, y32, y16);
  // blocks_2: alternate self-attn (mask=1) / cross-attn on y (mask=0, strict)
  run_layer(2, x32, x16, x16, 0, x32, x16);
  run_layer(3, x32, x16, y16, 1, x32, x16);
  run_layer(4, x32, x16, x16, 0, x32, x16);
  run_layer(5, x32, x16, y16, 1, out32, x16);  // final LN writes straight to d_out
}
